// Head_48524540510681
// MI455X (gfx1250) — matrix operand order repaired and checked by reading, not compiled, not run on hardware
//
#include <hip/hip_runtime.h>

// Problem constants (match reference)
#define BATCH 4
#define TLEN  4096
#define CDIM  1024
#define HSZ   64
#define BT    (BATCH * TLEN)

typedef __bf16 bf16;
typedef __attribute__((ext_vector_type(16))) __bf16 v16bf;
typedef __attribute__((ext_vector_type(8)))  float  v8f;

static __device__ __forceinline__ v8f wmma_bf16(v16bf a, v16bf b, v8f c) {
  // D = A(16x32 bf16) * B(32x16 bf16) + C(16x16 f32)
  return __builtin_amdgcn_wmma_f32_16x16x32_bf16(
      /*neg_a=*/false, a, /*neg_b=*/false, b,
      /*c_mod=*/(short)0, c, /*reuse_a=*/false, /*reuse_b=*/false);
}

// Async DMA: global -> LDS, 16 bytes per lane, tracked by ASYNCcnt.
static __device__ __forceinline__ void async_copy_b128(const void* gptr, void* lds_ptr) {
  // LDS logical offset = low 32 bits of the flat shared-aperture address.
  unsigned lds_off = (unsigned)(unsigned long long)lds_ptr;
  asm volatile("global_load_async_to_lds_b128 %0, %1, off"
               :: "v"(lds_off), "v"(gptr) : "memory");
}
static __device__ __forceinline__ void wait_async0() {
  asm volatile("s_wait_asynccnt 0" ::: "memory");
}

// ---------------------------------------------------------------------------
// Kernel 1: Q/K/V projection.  grid = (BT/128, 3), block = 256 (8 wave32)
// out[g] = bf16( x @ W_g + b_g ),  x:[BT,C] f32, W:[C,HS] f32 -> out:[BT,HS] bf16
// ---------------------------------------------------------------------------
__global__ __launch_bounds__(256)
void qkv_proj(const float* __restrict__ x,
              const float* __restrict__ Wq, const float* __restrict__ bq,
              const float* __restrict__ Wk, const float* __restrict__ bk,
              const float* __restrict__ Wv, const float* __restrict__ bv,
              bf16* __restrict__ Qbf, bf16* __restrict__ Kbf, bf16* __restrict__ Vbf)
{
  const float* W; const float* bias; bf16* out;
  if (blockIdx.y == 0)      { W = Wq; bias = bq; out = Qbf; }
  else if (blockIdx.y == 1) { W = Wk; bias = bk; out = Kbf; }
  else                      { W = Wv; bias = bv; out = Vbf; }

  // W slab, transposed: sWt[c][k] for k-slab of 128   (64*128 bf16 = 16 KB)
  __shared__ __align__(16) bf16 sWt[HSZ * 128];

  const int tid   = threadIdx.x;
  const int wave  = tid >> 5;
  const int lane  = tid & 31;
  const int n     = lane & 15;
  const int mbase = (lane < 16) ? 0 : 8;   // C/D row offset for this lane half
  const int aoff  = (lane < 16) ? 0 : 8;   // unused (A frag uses koff)
  const int koff  = (lane < 16) ? 0 : 16;  // B-frag K offset
  const int rowbase = blockIdx.x * 128 + wave * 16;
  const int arow    = rowbase + n;         // A-matrix row owned by this lane

  v8f acc[4];
  #pragma unroll
  for (int t = 0; t < 4; ++t)
    #pragma unroll
    for (int r = 0; r < 8; ++r) acc[t][r] = 0.0f;

  for (int k0 = 0; k0 < CDIM; k0 += 128) {
    __syncthreads();
    // Cooperative load of W[k0..k0+127][0..63], f32 -> bf16, transposed.
    #pragma unroll
    for (int e = 0; e < 4; ++e) {
      int lin = (e * 256 + tid) * 8;       // 0 .. 8191
      int kk  = lin >> 6;                  // 0 .. 127
      int c0  = lin & 63;                  // multiple of 8
      const float* wp = W + (k0 + kk) * HSZ + c0;
      #pragma unroll
      for (int j = 0; j < 8; ++j)
        sWt[(c0 + j) * 128 + kk] = (bf16)wp[j];
    }
    __syncthreads();

    #pragma unroll
    for (int kc = 0; kc < 128; kc += 32) {
      // A fragment: 16x32 bf16 from x (converted on the fly)
      v16bf a;
      const float* xp = x + arow * CDIM + k0 + kc;
      #pragma unroll
      for (int j = 0; j < 8; ++j) {
        a[j]     = (bf16)xp[koff + j];
        a[8 + j] = (bf16)xp[koff + 8 + j];
      }
      #pragma unroll
      for (int t = 0; t < 4; ++t) {
        v16bf bf;
        const bf16* bp = &sWt[(t * 16 + n) * 128 + kc + koff];
        #pragma unroll
        for (int j = 0; j < 16; ++j) bf[j] = bp[j];
        acc[t] = wmma_bf16(a, bf, acc[t]);
      }
    }
  }

  // Epilogue: add bias, convert to bf16, store
  #pragma unroll
  for (int t = 0; t < 4; ++t) {
    int col = t * 16 + n;
    float bb = bias[col];
    #pragma unroll
    for (int r = 0; r < 8; ++r) {
      int row = rowbase + mbase + r;
      out[row * HSZ + col] = (bf16)(acc[t][r] + bb);
    }
  }
}

// ---------------------------------------------------------------------------
// Kernel 2: causal flash attention.  grid = (T/128, B), block = 256 (8 wave32)
// Each wave owns 16 query rows; 32-key tiles double-buffered:
//   K: async global->LDS DMA (ASYNCcnt), row-major
//   V: global load early, VGPR-transpose ds-stores after compute
// ---------------------------------------------------------------------------
__global__ __launch_bounds__(256)
void flash_attn(const bf16* __restrict__ Qbf, const bf16* __restrict__ Kbf,
                const bf16* __restrict__ Vbf, float* __restrict__ outp)
{
  __shared__ __align__(16) bf16 sK [2][32 * HSZ];   // [key][feat]  2 x 4 KB
  __shared__ __align__(16) bf16 sVt[2][HSZ * 32];   // [feat][key]  2 x 4 KB
  __shared__ __align__(16) bf16 sP [8 * 16 * 32];   // per-wave P scratch 8 KB

  const int tid   = threadIdx.x;
  const int wave  = tid >> 5;
  const int lane  = tid & 31;
  const int n     = lane & 15;
  const int mbase = (lane < 16) ? 0 : 8;
  const int aoff  = (lane < 16) ? 0 : 8;
  const int koff  = (lane < 16) ? 0 : 16;

  const int  b         = blockIdx.y;
  const int  qbase     = blockIdx.x * 128;
  const int  qrow0     = qbase + wave * 16;
  const int  wave_qmax = qrow0 + 15;
  const long rbase     = (long)b * TLEN;            // row offset within [BT]

  // Cooperative-stage coordinates: thread -> (key row, feature chunk of 8)
  const int key = tid >> 3;
  const int c0  = (tid & 7) * 8;

  // Q fragments for this wave's 16 rows (K = 0..31 and 32..63)
  v16bf qa0, qa1;
  {
    const bf16* qp = Qbf + (rbase + qrow0 + n) * HSZ;
    #pragma unroll
    for (int j = 0; j < 8; ++j) {
      qa0[j]     = qp[koff + j];
      qa0[8 + j] = qp[koff + 8 + j];
      qa1[j]     = qp[32 + koff + j];
      qa1[8 + j] = qp[32 + koff + 8 + j];
    }
  }

  v8f acc[4];
  float mrun[8], lrun[8];
  #pragma unroll
  for (int t = 0; t < 4; ++t)
    #pragma unroll
    for (int r = 0; r < 8; ++r) acc[t][r] = 0.0f;
  #pragma unroll
  for (int r = 0; r < 8; ++r) { mrun[r] = -1e30f; lrun[r] = 0.0f; }

  const float scale  = 0.03125f;                 // C^-0.5 = 1/32 (faithful to ref)
  const int   ntiles = blockIdx.x * 4 + 4;       // keys 0 .. qbase+127
  bf16* pbase = sP + wave * 512;

  // ---- prologue: stage tile 0 into buffer 0 ----
  {
    async_copy_b128(Kbf + (rbase + key) * HSZ + c0, &sK[0][key * HSZ + c0]);
    uint4 vv = *(const uint4*)(Vbf + (rbase + key) * HSZ + c0);
    const bf16* ve = (const bf16*)&vv;
    #pragma unroll
    for (int j = 0; j < 8; ++j)
      sVt[0][(c0 + j) * 32 + key] = ve[j];
  }

  int cur = 0;
  for (int it = 0; it < ntiles; ++it) {
    const int kbase = it * 32;

    wait_async0();          // this wave's async K DMA for buffer `cur` done
    __syncthreads();        // all waves' DMAs + V transpose stores visible

    const int  nxt = cur ^ 1;
    const bool pf  = (it + 1 < ntiles);
    uint4 vv;
    if (pf) {
      // issue next tile's K DMA + V global load now; overlap with compute
      async_copy_b128(Kbf + (rbase + kbase + 32 + key) * HSZ + c0,
                      &sK[nxt][key * HSZ + c0]);
      vv = *(const uint4*)(Vbf + (rbase + kbase + 32 + key) * HSZ + c0);
    }

    if (kbase <= wave_qmax) {            // wave-uniform causal skip (EXEC stays full)
      const bf16* sKc  = sK[cur];
      const bf16* sVtc = sVt[cur];

      // ---- S = Q @ K^T : two 16x16 tiles, K-dim = 64 (2 WMMA chunks each) ----
      v8f s0, s1;
      #pragma unroll
      for (int r = 0; r < 8; ++r) { s0[r] = 0.0f; s1[r] = 0.0f; }
      {
        v16bf kb;
        const bf16* kp;
        kp = &sKc[n * HSZ + koff];
        #pragma unroll
        for (int j = 0; j < 16; ++j) kb[j] = kp[j];
        s0 = wmma_bf16(qa0, kb, s0);
        kp = &sKc[n * HSZ + 32 + koff];
        #pragma unroll
        for (int j = 0; j < 16; ++j) kb[j] = kp[j];
        s0 = wmma_bf16(qa1, kb, s0);
        kp = &sKc[(16 + n) * HSZ + koff];
        #pragma unroll
        for (int j = 0; j < 16; ++j) kb[j] = kp[j];
        s1 = wmma_bf16(qa0, kb, s1);
        kp = &sKc[(16 + n) * HSZ + 32 + koff];
        #pragma unroll
        for (int j = 0; j < 16; ++j) kb[j] = kp[j];
        s1 = wmma_bf16(qa1, kb, s1);
      }

      // ---- online softmax (per-row stats across the 16-lane N group) ----
      #pragma unroll
      for (int r = 0; r < 8; ++r) {
        int row  = qrow0 + mbase + r;
        int col0 = kbase + n;
        int col1 = kbase + 16 + n;
        float a0 = (col0 <= row) ? s0[r] * scale : -1e30f;
        float a1 = (col1 <= row) ? s1[r] * scale : -1e30f;
        float mx = fmaxf(a0, a1);
        mx = fmaxf(mx, __shfl_xor(mx, 1, 32));
        mx = fmaxf(mx, __shfl_xor(mx, 2, 32));
        mx = fmaxf(mx, __shfl_xor(mx, 4, 32));
        mx = fmaxf(mx, __shfl_xor(mx, 8, 32));
        float mnew = fmaxf(mrun[r], mx);
        float corr = __expf(mrun[r] - mnew);
        float p0 = __expf(a0 - mnew);
        float p1 = __expf(a1 - mnew);
        float ps = p0 + p1;
        ps += __shfl_xor(ps, 1, 32);
        ps += __shfl_xor(ps, 2, 32);
        ps += __shfl_xor(ps, 4, 32);
        ps += __shfl_xor(ps, 8, 32);
        lrun[r] = lrun[r] * corr + ps;
        mrun[r] = mnew;
        pbase[(mbase + r) * 32 + n]      = (bf16)p0;
        pbase[(mbase + r) * 32 + 16 + n] = (bf16)p1;
        acc[0][r] *= corr; acc[1][r] *= corr; acc[2][r] *= corr; acc[3][r] *= corr;
      }

      // Re-layout P (C-layout -> A-fragment) through per-wave LDS scratch.
      asm volatile("s_wait_dscnt 0" ::: "memory");
      v16bf pa;
      {
        const bf16* pp = pbase + (lane & 15) * 32;
        #pragma unroll
        for (int j = 0; j < 8; ++j) {
          pa[j]     = pp[koff + j];
          pa[8 + j] = pp[koff + 8 + j];
        }
      }

      // ---- O += P @ V  (K = 32 keys, 4 N-tiles over HS=64) ----
      #pragma unroll
      for (int t = 0; t < 4; ++t) {
        v16bf vb;
        const bf16* vp = &sVtc[(t * 16 + n) * 32 + koff];
        #pragma unroll
        for (int j = 0; j < 16; ++j) vb[j] = vp[j];
        acc[t] = wmma_bf16(pa, vb, acc[t]);
      }
    }

    if (pf) {
      // V transpose-store into the next buffer (after compute, before barrier)
      const bf16* ve = (const bf16*)&vv;
      #pragma unroll
      for (int j = 0; j < 8; ++j)
        sVt[nxt][(c0 + j) * 32 + key] = ve[j];
    }
    cur = nxt;
  }

  // ---- normalize and store f32 output ----
  #pragma unroll
  for (int t = 0; t < 4; ++t) {
    int col = t * 16 + n;
    #pragma unroll
    for (int r = 0; r < 8; ++r) {
      int row = qrow0 + mbase + r;
      outp[(rbase + row) * HSZ + col] = acc[t][r] / lrun[r];
    }
  }
}

// ---------------------------------------------------------------------------
extern "C" void kernel_launch(void* const* d_in, const int* in_sizes, int n_in,
                              void* d_out, int out_size, void* d_ws, size_t ws_size,
                              hipStream_t stream) {
  const float* x  = (const float*)d_in[0];
  const float* Wq = (const float*)d_in[1];
  const float* bq = (const float*)d_in[2];
  const float* Wk = (const float*)d_in[3];
  const float* bk = (const float*)d_in[4];
  const float* Wv = (const float*)d_in[5];
  const float* bv = (const float*)d_in[6];

  bf16* Qbf = (bf16*)d_ws;                       // BT*HS bf16 = 2 MB
  bf16* Kbf = Qbf + (size_t)BT * HSZ;            // 2 MB
  bf16* Vbf = Kbf + (size_t)BT * HSZ;            // 2 MB
  float* outp = (float*)d_out;

  dim3 gq(BT / 128, 3);
  qkv_proj<<<gq, 256, 0, stream>>>(x, Wq, bq, Wk, bk, Wv, bv, Qbf, Kbf, Vbf);

  dim3 gf(TLEN / 128, BATCH);
  flash_attn<<<gf, 256, 0, stream>>>(Qbf, Kbf, Vbf, outp);
}
